// TransformerDecoder_13804024890060
// MI455X (gfx1250) — compile-verified, matrix operand order checked by reading, not run on hardware
//
#include <hip/hip_runtime.h>

// ---------------------------------------------------------------------------
// Model dims (fixed by reference)
// ---------------------------------------------------------------------------
#define LL   6
#define DD   1024
#define HH   16
#define DHH  64
#define DFF  4096
#define VV   32000
#define SS   512
#define MM   1024
#define BB   8

typedef __attribute__((ext_vector_type(16))) _Float16 v16h;
typedef __attribute__((ext_vector_type(8)))  _Float16 v8h;
typedef __attribute__((ext_vector_type(8)))  float    v8f;
typedef __attribute__((ext_vector_type(4)))  float    v4f;

// Convert a generic pointer to a __shared__ object into its 32-bit LDS byte
// offset (what GLOBAL_LOAD_ASYNC_TO_LDS wants in its VDST operand).
#define LDS_OFF(p) ((unsigned)(uintptr_t)(const __attribute__((address_space(3))) void*)(p))

// Issue one 16-byte async global->LDS copy (tracked with ASYNCcnt).
static __device__ __forceinline__ void async_b128(unsigned lds_byte_off, const float* gaddr) {
  asm volatile("global_load_async_to_lds_b128 %0, %1, off"
               :: "v"(lds_byte_off), "v"(gaddr) : "memory");
}

// ---------------------------------------------------------------------------
// Generic GEMM: C[M,N] = A[M,K] @ B  (+bias) (+relu)
//   BT (template): B stored [N,K] row-major (use B^T), else [K,N].
//   flags bit0: relu
// Requires M%128==0, N%128==0, K%32==0 (true at all call sites).
// Block = 256 threads (8 waves), tile 128x128, K-step 32.
// Double-buffered fp32 LDS tiles filled by async copies; f32->f16 conversion
// at fragment-build time; v_wmma_f32_16x16x32_f16 compute, f32 accumulate.
// ---------------------------------------------------------------------------
#define GBM 128
#define GBN 128
#define GBK 32

// Build a 16x32 f16 A-style fragment from a 128x32 fp32 LDS tile whose 16B
// chunks are XOR-swizzled: chunk c of row r lives at (c ^ (r & 7)).
// Frag layout (lane = l): m = l&15, halves 0..7 = K klo..klo+7,
// halves 8..15 = K klo+16..klo+23, klo = 8*(l>=16).
static __device__ __forceinline__ v16h frag_from_swz(const float* tile, int r, int klo) {
  const int sw = r & 7;
  const int cb = klo >> 2;   // 0 or 2
  const float* rp = tile + r * GBK;
  v4f a0 = *(const v4f*)(rp + (((cb + 0) ^ sw) << 2));
  v4f a1 = *(const v4f*)(rp + (((cb + 1) ^ sw) << 2));
  v4f b0 = *(const v4f*)(rp + (((cb + 4) ^ sw) << 2));
  v4f b1 = *(const v4f*)(rp + (((cb + 5) ^ sw) << 2));
  v16h f;
#pragma unroll
  for (int j = 0; j < 4; ++j) {
    f[j]      = (_Float16)a0[j];
    f[4 + j]  = (_Float16)a1[j];
    f[8 + j]  = (_Float16)b0[j];
    f[12 + j] = (_Float16)b1[j];
  }
  return f;
}

template <bool BT>
__global__ __launch_bounds__(256) void gemm_kernel(
    const float* __restrict__ A, const float* __restrict__ Bm,
    const float* __restrict__ bias, float* __restrict__ C,
    int Mdim, int Ndim, int Kdim, int flags)
{
  // fp32 tiles, double-buffered: 2 * (16KB + 16KB) = 64KB LDS
  __shared__ float Asf[2][GBM * GBK];
  __shared__ float Bsf[2][GBM * GBK];   // BT: [n][k] swizzled; !BT: [k][n] linear

  const int tid  = threadIdx.x;
  const int lane = tid & 31;
  const int wid  = tid >> 5;
  const int wm   = wid >> 2;        // 0..1  -> 64 rows each
  const int wn   = wid & 3;         // 0..3  -> 32 cols each
  const int bm0  = blockIdx.y * GBM;
  const int bn0  = blockIdx.x * GBN;
  const int ml   = lane & 15;
  const int hi   = lane >> 4;
  const int klo  = hi << 3;         // frag K base (0 or 8)

  const unsigned ldsA[2] = { LDS_OFF(&Asf[0][0]), LDS_OFF(&Asf[1][0]) };
  const unsigned ldsB[2] = { LDS_OFF(&Bsf[0][0]), LDS_OFF(&Bsf[1][0]) };

  auto stage = [&](int kb, int buf) {
#pragma unroll
    for (int i = 0; i < 4; ++i) {
      const int c   = tid + i * 256;
      const int row = c >> 3;
      const int ck  = c & 7;
      const float* g = A + (size_t)(bm0 + row) * Kdim + kb + ck * 4;
      async_b128(ldsA[buf] + (unsigned)((row * GBK + ((ck ^ (row & 7)) << 2)) * 4), g);
    }
    if (BT) {
#pragma unroll
      for (int i = 0; i < 4; ++i) {
        const int c   = tid + i * 256;
        const int row = c >> 3;               // n index
        const int ck  = c & 7;
        const float* g = Bm + (size_t)(bn0 + row) * Kdim + kb + ck * 4;
        async_b128(ldsB[buf] + (unsigned)((row * GBK + ((ck ^ (row & 7)) << 2)) * 4), g);
      }
    } else {
#pragma unroll
      for (int i = 0; i < 4; ++i) {
        const int c   = tid + i * 256;
        const int row = c >> 5;               // k index (0..31)
        const int ck  = c & 31;
        const float* g = Bm + (size_t)(kb + row) * Ndim + bn0 + ck * 4;
        async_b128(ldsB[buf] + (unsigned)((row * GBN + ck * 4) * 4), g);
      }
    }
  };

  v8f acc[4][2];
#pragma unroll
  for (int im = 0; im < 4; ++im)
#pragma unroll
    for (int in = 0; in < 2; ++in) acc[im][in] = (v8f)0.f;

  const int nk = Kdim / GBK;
  stage(0, 0);

  for (int kb = 0; kb < nk; ++kb) {
    const int cur = kb & 1;
    asm volatile("s_wait_asynccnt 0x0" ::: "memory");
    __syncthreads();                       // tile kb resident; buf cur^1 free
    if (kb + 1 < nk) stage((kb + 1) * GBK, cur ^ 1);

    v16h af[4], bf[2];
#pragma unroll
    for (int im = 0; im < 4; ++im)
      af[im] = frag_from_swz(&Asf[cur][0], wm * 64 + im * 16 + ml, klo);

    if (BT) {
#pragma unroll
      for (int in = 0; in < 2; ++in)
        bf[in] = frag_from_swz(&Bsf[cur][0], wn * 32 + in * 16 + ml, klo);
    } else {
#pragma unroll
      for (int in = 0; in < 2; ++in) {
        const int n = wn * 32 + in * 16 + ml;
        const float* bp = &Bsf[cur][0];
#pragma unroll
        for (int vv = 0; vv < 8; ++vv) {
          const int kk = ((vv >> 2) << 4) + klo + ((vv & 3) << 1);
          bf[in][2 * vv]     = (_Float16)bp[kk * GBN + n];
          bf[in][2 * vv + 1] = (_Float16)bp[(kk + 1) * GBN + n];
        }
      }
    }

#pragma unroll
    for (int im = 0; im < 4; ++im)
#pragma unroll
      for (int in = 0; in < 2; ++in)
        acc[im][in] = __builtin_amdgcn_wmma_f32_16x16x32_f16(
            false, af[im], false, bf[in], (short)0, acc[im][in], false, false);
  }

  // epilogue: C/D layout -> lane = N, vgpr r = M row (+8 for upper half-wave)
  const bool relu = (flags & 1) != 0;
#pragma unroll
  for (int im = 0; im < 4; ++im)
#pragma unroll
    for (int in = 0; in < 2; ++in)
#pragma unroll
      for (int r = 0; r < 8; ++r) {
        const int gm = bm0 + wm * 64 + im * 16 + r + (hi << 3);
        const int gn = bn0 + wn * 32 + in * 16 + ml;
        float vv = acc[im][in][r];
        if (bias) vv += bias[gn];
        if (relu) vv = fmaxf(vv, 0.f);
        C[(size_t)gm * Ndim + gn] = vv;
      }
}

// ---------------------------------------------------------------------------
// Flash-style attention, workgroup version.
// Block = 256 threads (8 waves); each wave owns a 16-row query tile, the block
// covers 128 query rows of one (b,h). K/V 32-key tiles are async-copied into
// double-buffered LDS once per block and shared by all 8 waves.
// q: [B,S,H*DH], k/v: [B,Lk,H*DH], o: [B,S,H*DH].  Online softmax, f32 state.
// ---------------------------------------------------------------------------
#define LDK 40

// float index of (key, d) inside a 32x64 K/V LDS tile with 16B-chunk swizzle
// (chunk ch of row `key` stored at ch ^ (key & 15)).
static __device__ __forceinline__ int kv_off(int key, int d) {
  return key * 64 + (((d >> 2) ^ (key & 15)) << 2) + (d & 3);
}

// B-style fragment for K: lane n = key (within 16-key subtile), contraction =
// d in [dbase, dbase+32).
static __device__ __forceinline__ v16h kfrag(const float* tile, int n, int dbase, int klo) {
  v4f a0 = *(const v4f*)(tile + kv_off(n, dbase + klo));
  v4f a1 = *(const v4f*)(tile + kv_off(n, dbase + klo + 4));
  v4f b0 = *(const v4f*)(tile + kv_off(n, dbase + klo + 16));
  v4f b1 = *(const v4f*)(tile + kv_off(n, dbase + klo + 20));
  v16h f;
#pragma unroll
  for (int j = 0; j < 4; ++j) {
    f[j]      = (_Float16)a0[j];
    f[4 + j]  = (_Float16)a1[j];
    f[8 + j]  = (_Float16)b0[j];
    f[12 + j] = (_Float16)b1[j];
  }
  return f;
}

__global__ __launch_bounds__(256) void attn_kernel(
    const float* __restrict__ q, const float* __restrict__ k,
    const float* __restrict__ v, float* __restrict__ o,
    int Lk, int causal)
{
  __shared__ float Ks[2][32 * 64];           // 2 x 8KB
  __shared__ float Vs[2][32 * 64];           // 2 x 8KB
  __shared__ _Float16 Plds[8][16 * LDK];     // per-wave P staging (10KB)

  const int tid  = threadIdx.x;
  const int lane = tid & 31;
  const int wid  = tid >> 5;
  const int ml = lane & 15, hi = lane >> 4;
  const int klo = hi << 3;
  const int b = blockIdx.x >> 4, h = blockIdx.x & (HH - 1);
  const int qb = blockIdx.y;
  const int qt = qb * 8 + wid;               // this wave's 16-row query tile

  const float* kvk = k + (size_t)b * Lk * DD + h * DHH;
  const float* kvv = v + (size_t)b * Lk * DD + h * DHH;
  const unsigned ldsK[2] = { LDS_OFF(&Ks[0][0]), LDS_OFF(&Ks[1][0]) };
  const unsigned ldsV[2] = { LDS_OFF(&Vs[0][0]), LDS_OFF(&Vs[1][0]) };

  auto stage = [&](int kt, int buf) {
    // one 32x64 fp32 tile = 512 16B chunks; 256 threads x 2 chunks x {K,V}
#pragma unroll
    for (int i = 0; i < 2; ++i) {
      const int c   = tid + i * 256;
      const int key = c >> 4;
      const int dc  = c & 15;                // 16B chunk within row
      const unsigned lo = (unsigned)((key * 64 + ((dc ^ (key & 15)) << 2)) * 4);
      const size_t   go = (size_t)(kt * 32 + key) * DD + dc * 4;
      async_b128(ldsK[buf] + lo, kvk + go);
      async_b128(ldsV[buf] + lo, kvv + go);
    }
  };

  // Q fragments (16x64 -> two 16x32 A-frags), loaded once from global
  v16h qa[2];
  {
    const float* qp = q + ((size_t)(b * SS + qt * 16 + ml)) * DD + h * DHH;
#pragma unroll
    for (int c = 0; c < 2; ++c)
#pragma unroll
      for (int vv2 = 0; vv2 < 8; ++vv2) {
        const int k0 = c * 32 + ((vv2 >> 2) << 4) + klo + ((vv2 & 3) << 1);
        qa[c][2 * vv2]     = (_Float16)qp[k0];
        qa[c][2 * vv2 + 1] = (_Float16)qp[k0 + 1];
      }
  }

  float mrun[8], lrun[8];
  v8f oacc[4];
#pragma unroll
  for (int r = 0; r < 8; ++r) { mrun[r] = -1e30f; lrun[r] = 0.f; }
#pragma unroll
  for (int t = 0; t < 4; ++t) oacc[t] = (v8f)0.f;

  // tiles the whole block must stage vs tiles this wave must compute
  const int nkt_blk = causal ? (4 * qb + 4) : (Lk / 32);
  const int my_nkt  = __builtin_amdgcn_readfirstlane(
      causal ? (qt * 16 + 16 + 31) / 32 : (Lk / 32));

  stage(0, 0);

  for (int kt = 0; kt < nkt_blk; ++kt) {
    const int cur = kt & 1;
    asm volatile("s_wait_asynccnt 0x0" ::: "memory");
    __syncthreads();                        // tile kt resident; buf cur^1 free
    if (kt + 1 < nkt_blk) stage(kt + 1, cur ^ 1);

    if (kt < my_nkt) {                      // scalar (wave-uniform) guard
      const float* Kt = &Ks[cur][0];
      const float* Vt = &Vs[cur][0];

      // scores: 16x32 tile = two 16x16 C-frags
      v8f sc[2];
#pragma unroll
      for (int j = 0; j < 2; ++j) {
        const int n = j * 16 + ml;
        v16h kb0 = kfrag(Kt, n, 0,  klo);
        v16h kb1 = kfrag(Kt, n, 32, klo);
        v8f z = (v8f)0.f;
        z = __builtin_amdgcn_wmma_f32_16x16x32_f16(false, qa[0], false, kb0, (short)0, z, false, false);
        z = __builtin_amdgcn_wmma_f32_16x16x32_f16(false, qa[1], false, kb1, (short)0, z, false, false);
        sc[j] = z;
      }

      // scale + causal mask + online softmax (row = r + 8*hi, 16-lane groups)
      float pj[2][8];
#pragma unroll
      for (int r = 0; r < 8; ++r) {
        const int gq = qt * 16 + r + (hi << 3);
        float s0 = sc[0][r] * 0.125f;
        float s1 = sc[1][r] * 0.125f;
        if (causal) {
          const int gk0 = kt * 32 + ml;
          if (gk0 > gq)      s0 = -1e9f;
          if (gk0 + 16 > gq) s1 = -1e9f;
        }
        float rm = fmaxf(s0, s1);
#pragma unroll
        for (int off = 8; off > 0; off >>= 1)
          rm = fmaxf(rm, __shfl_xor(rm, off, 16));
        const float mnew  = fmaxf(mrun[r], rm);
        const float alpha = __expf(mrun[r] - mnew);
        const float p0 = __expf(s0 - mnew);
        const float p1 = __expf(s1 - mnew);
        float rs = p0 + p1;
#pragma unroll
        for (int off = 8; off > 0; off >>= 1)
          rs += __shfl_xor(rs, off, 16);
        lrun[r] = lrun[r] * alpha + rs;
        mrun[r] = mnew;
        pj[0][r] = p0; pj[1][r] = p1;
#pragma unroll
        for (int t = 0; t < 4; ++t) oacc[t][r] *= alpha;
      }

      // P: C-layout -> A-layout via per-wave LDS (in-order DS within one wave)
      _Float16* pw = &Plds[wid][0];
#pragma unroll
      for (int j = 0; j < 2; ++j)
#pragma unroll
        for (int r = 0; r < 8; ++r)
          pw[(r + (hi << 3)) * LDK + j * 16 + ml] = (_Float16)pj[j][r];
      asm volatile("" ::: "memory");
      v16h pa;
      {
        const _Float16* pp = &pw[ml * LDK + klo];
        *(v8h*)&pa       = *(const v8h*)pp;
        *((v8h*)&pa + 1) = *(const v8h*)(pp + 16);
      }
      asm volatile("" ::: "memory");

      // o += P (16x32) @ V (32x64)
#pragma unroll
      for (int t = 0; t < 4; ++t) {
        const int d = t * 16 + ml;
        v16h vbf;
#pragma unroll
        for (int vv2 = 0; vv2 < 8; ++vv2) {
          const int kk0 = ((vv2 >> 2) << 4) + klo + ((vv2 & 3) << 1);
          vbf[2 * vv2]     = (_Float16)Vt[kv_off(kk0, d)];
          vbf[2 * vv2 + 1] = (_Float16)Vt[kv_off(kk0 + 1, d)];
        }
        oacc[t] = __builtin_amdgcn_wmma_f32_16x16x32_f16(
            false, pa, false, vbf, (short)0, oacc[t], false, false);
      }
    }
  }

  // normalize + store
#pragma unroll
  for (int t = 0; t < 4; ++t)
#pragma unroll
    for (int r = 0; r < 8; ++r) {
      const int gq = qt * 16 + r + (hi << 3);
      o[((size_t)(b * SS + gq)) * DD + h * DHH + t * 16 + ml] = oacc[t][r] / lrun[r];
    }
}

// ---------------------------------------------------------------------------
// out[row] = LN(xin[row] (+ resid[row])) * sc + bi       (row length DD=1024)
// ---------------------------------------------------------------------------
__global__ __launch_bounds__(256) void add_ln_kernel(
    const float* __restrict__ xin, const float* __restrict__ resid,
    const float* __restrict__ sc, const float* __restrict__ bi,
    float* __restrict__ out)
{
  __shared__ float red[256];
  const int row = blockIdx.x, tid = threadIdx.x;
  float vals[4];
  float s = 0.f;
#pragma unroll
  for (int j = 0; j < 4; ++j) {
    const int d = tid + j * 256;
    float x = xin[(size_t)row * DD + d];
    if (resid) x += resid[(size_t)row * DD + d];
    vals[j] = x; s += x;
  }
  red[tid] = s; __syncthreads();
  for (int o2 = 128; o2 > 0; o2 >>= 1) {
    if (tid < o2) red[tid] += red[tid + o2];
    __syncthreads();
  }
  const float mean = red[0] * (1.f / DD);
  __syncthreads();
  float sq = 0.f;
#pragma unroll
  for (int j = 0; j < 4; ++j) { const float d2 = vals[j] - mean; sq += d2 * d2; }
  red[tid] = sq; __syncthreads();
  for (int o2 = 128; o2 > 0; o2 >>= 1) {
    if (tid < o2) red[tid] += red[tid + o2];
    __syncthreads();
  }
  const float inv = rsqrtf(red[0] * (1.f / DD) + 1e-5f);
#pragma unroll
  for (int j = 0; j < 4; ++j) {
    const int d = tid + j * 256;
    out[(size_t)row * DD + d] = (vals[j] - mean) * inv * sc[d] + bi[d];
  }
}

// ---------------------------------------------------------------------------
// x[b,s,:] = input_embed[seq[b,s],:] + pos_embed[s,:]
// ---------------------------------------------------------------------------
__global__ void embed_kernel(const int* __restrict__ seq,
                             const float* __restrict__ emb,
                             const float* __restrict__ pos,
                             float* __restrict__ out)
{
  const size_t i = (size_t)blockIdx.x * blockDim.x + threadIdx.x;
  if (i >= (size_t)BB * SS * DD) return;
  const int d = (int)(i % DD);
  const size_t bs = i / DD;
  const int s = (int)(bs % SS);
  const int tok = seq[bs];
  out[i] = emb[(size_t)tok * DD + d] + pos[(size_t)s * DD + d];
}

// ---------------------------------------------------------------------------
// Orchestration
// ---------------------------------------------------------------------------
extern "C" void kernel_launch(void* const* d_in, const int* in_sizes, int n_in,
                              void* d_out, int out_size, void* d_ws, size_t ws_size,
                              hipStream_t stream) {
  (void)in_sizes; (void)n_in; (void)out_size; (void)ws_size;

  const float* encoded      = (const float*)d_in[0];
  const int*   seq          = (const int*)  d_in[1];
  const float* input_embed  = (const float*)d_in[2];
  const float* output_embed = (const float*)d_in[3];
  const float* output_bias  = (const float*)d_in[4];
  const float* pos_embed    = (const float*)d_in[5];
  const float* Wq_  = (const float*)d_in[6];
  const float* Wk_  = (const float*)d_in[7];
  const float* Wv_  = (const float*)d_in[8];
  const float* Wo_  = (const float*)d_in[9];
  const float* Wqc_ = (const float*)d_in[10];
  const float* Wkc_ = (const float*)d_in[11];
  const float* Wvc_ = (const float*)d_in[12];
  const float* Woc_ = (const float*)d_in[13];
  const float* W1_  = (const float*)d_in[14];
  const float* b1_  = (const float*)d_in[15];
  const float* W2_  = (const float*)d_in[16];
  const float* b2_  = (const float*)d_in[17];
  const float* ln1s = (const float*)d_in[18];
  const float* ln1b = (const float*)d_in[19];
  const float* ln2s = (const float*)d_in[20];
  const float* ln2b = (const float*)d_in[21];
  const float* ln3s = (const float*)d_in[22];
  const float* ln3b = (const float*)d_in[23];
  const float* lnfs = (const float*)d_in[24];
  const float* lnfb = (const float*)d_in[25];

  // workspace layout (floats)
  const size_t XSZ  = (size_t)BB * SS * DD;   // 4M
  const size_t KVSZ = (size_t)BB * MM * DD;   // 8M
  float* ws = (float*)d_ws;
  float* x  = ws;
  float* a  = x  + XSZ;
  float* t  = a  + XSZ;
  float* kb = t  + XSZ;
  float* vb = kb + KVSZ;
  float* hb = vb + KVSZ;

  auto gemm = [&](const float* A, const float* Bm, const float* bias, float* C,
                  int Md, int Nd, int Kd, int flags) {
    dim3 g(Nd / GBN, Md / GBM);
    if (flags & 2)
      gemm_kernel<true><<<g, 256, 0, stream>>>(A, Bm, bias, C, Md, Nd, Kd, flags);
    else
      gemm_kernel<false><<<g, 256, 0, stream>>>(A, Bm, bias, C, Md, Nd, Kd, flags);
  };

  const int NT = BB * SS;  // 4096 token rows

  // embedding + positional
  {
    const size_t n = (size_t)BB * SS * DD;
    embed_kernel<<<(unsigned)((n + 255) / 256), 256, 0, stream>>>(seq, input_embed, pos_embed, x);
  }

  const dim3 agrid(BB * HH, SS / 128);   // 8-wave blocks, 128 q rows each

  for (int i = 0; i < LL; ++i) {
    const size_t wo = (size_t)i * DD * DD;
    // ---- self-attention block
    gemm(x, Wq_ + wo, nullptr, a,  NT, DD, DD, 0);
    gemm(x, Wk_ + wo, nullptr, kb, NT, DD, DD, 0);
    gemm(x, Wv_ + wo, nullptr, vb, NT, DD, DD, 0);
    attn_kernel<<<agrid, 256, 0, stream>>>(a, kb, vb, t, SS, 1);
    gemm(t, Wo_ + wo, nullptr, a, NT, DD, DD, 0);
    add_ln_kernel<<<NT, 256, 0, stream>>>(x, a, ln1s + (size_t)i * DD, ln1b + (size_t)i * DD, x);
    // ---- cross-attention block
    gemm(x,       Wqc_ + wo, nullptr, a,  NT,      DD, DD, 0);
    gemm(encoded, Wkc_ + wo, nullptr, kb, BB * MM, DD, DD, 0);
    gemm(encoded, Wvc_ + wo, nullptr, vb, BB * MM, DD, DD, 0);
    attn_kernel<<<agrid, 256, 0, stream>>>(a, kb, vb, t, MM, 0);
    gemm(t, Woc_ + wo, nullptr, a, NT, DD, DD, 0);
    add_ln_kernel<<<NT, 256, 0, stream>>>(x, a, ln2s + (size_t)i * DD, ln2b + (size_t)i * DD, x);
    // ---- FFN block
    gemm(x,  W1_ + (size_t)i * DD * DFF, b1_ + (size_t)i * DFF, hb, NT, DFF, DD, 1 /*relu*/);
    gemm(hb, W2_ + (size_t)i * DFF * DD, b2_ + (size_t)i * DD,  a,  NT, DD, DFF, 0);
    add_ln_kernel<<<NT, 256, 0, stream>>>(x, a, ln3s + (size_t)i * DD, ln3b + (size_t)i * DD, x);
  }

  // final norm + logits (B^T gemm against output_embed [V,D])
  add_ln_kernel<<<NT, 256, 0, stream>>>(x, nullptr, lnfs, lnfb, t);
  gemm(t, output_embed, output_bias, (float*)d_out, NT, VV, DD, 2 /*bt*/);
}
